// Model_36833639531249
// MI455X (gfx1250) — compile-verified, matrix operand order checked by reading, not run on hardware
//
#include <hip/hip_runtime.h>
#include <hip/hip_bf16.h>
#include <hip/hip_fp16.h>
#include <stdint.h>

typedef __attribute__((ext_vector_type(16))) _Float16 v16h;
typedef __attribute__((ext_vector_type(8)))  _Float16 v8h;
typedef __attribute__((ext_vector_type(4)))  _Float16 v4h;
typedef __attribute__((ext_vector_type(8)))  float    v8f;

#define B_    32
#define H_    128
#define W0_   2048
#define W1_   2017
#define W2_   1954
#define W3_   1947
#define W1S_  2024   // planar W1 padded to multiple of 8 halves (16B rows)
#define W2S_  1960   // planar W2 padded to multiple of 8 halves
#define NEMB  512
#define DEMB  128
#define NROWS (B_ * W3_)               // 62304
#define YSIZE ((size_t)B_ * H_ * W0_)  // 8388608

// ---------------------------------------------------------------------------
// Zero accumulators (sse + histogram). Runs every launch (deterministic).
// ---------------------------------------------------------------------------
__global__ void init_accum(float* sse, int* hist) {
    int t = threadIdx.x;
    if (t == 0) *sse = 0.0f;
    if (t < NEMB) hist[t] = 0;
}

// ---------------------------------------------------------------------------
// Direct small 1-D conv (along W), fp32 math, LDS row staging.
// in layout: planar [b][IC][h][WinS].
// out: OILV ? channel-interleaved [b][h][WoutS][OC] (vector store per pos)
//            : planar [b][OC][h][WoutS]
// ---------------------------------------------------------------------------
template<typename InT, typename OutT, int IC, int OC, int KW, bool TR, bool RELU,
         bool OILV>
__global__ __launch_bounds__(256)
void conv_direct(const InT* __restrict__ in, const float* __restrict__ wg,
                 const float* __restrict__ bias, OutT* __restrict__ out,
                 int Win, int WinS, int Wout, int WoutS)
{
    constexpr int WIN = 256 + KW - 1;
    __shared__ float s_in[IC * WIN];
    __shared__ float s_w[OC * IC * KW];
    __shared__ float s_b[OC];

    const int tid = threadIdx.x;
    const int row = blockIdx.y;
    const int b = row / H_, h = row % H_;
    const int wb = blockIdx.x * 256;

    for (int i = tid; i < IC * WIN; i += 256) {
        int ch = i / WIN, t = i % WIN;
        int wsrc = TR ? (wb - (KW - 1) + t) : (wb + t);
        float v = 0.0f;
        if (wsrc >= 0 && wsrc < Win)
            v = (float)in[((size_t)(b * IC + ch) * H_ + h) * (size_t)WinS + wsrc];
        s_in[i] = v;
    }
    for (int i = tid; i < OC * IC * KW; i += 256) {
        int o = i / (IC * KW); int r = i % (IC * KW);
        int ch = r / KW; int tap = r % KW;
        s_w[i] = TR ? wg[(ch * OC + o) * KW + tap] : wg[i];
    }
    if (tid < OC) s_b[tid] = bias[tid];
    __syncthreads();

    const int w = wb + tid;
    if (w < Wout) {
        float acc[OC];
        #pragma unroll
        for (int o = 0; o < OC; ++o) acc[o] = s_b[o];
        #pragma unroll
        for (int ch = 0; ch < IC; ++ch) {
            #pragma unroll 8
            for (int tap = 0; tap < KW; ++tap) {
                int t = TR ? (tid + (KW - 1) - tap) : (tid + tap);
                float xv = s_in[ch * WIN + t];
                #pragma unroll
                for (int o = 0; o < OC; ++o)
                    acc[o] = fmaf(xv, s_w[(o * IC + ch) * KW + tap], acc[o]);
            }
        }
        if (OILV) {   // interleaved: one aligned vector store per position
            size_t base = (((size_t)b * H_ + h) * (size_t)WoutS + w) * OC;
            if (OC == 8) {
                v8h ov;
                #pragma unroll
                for (int o = 0; o < OC; ++o) {
                    float v = acc[o];
                    if (RELU) v = fmaxf(v, 0.0f);
                    ov[o & 7] = (_Float16)v;
                }
                *(v8h*)&out[base] = *(v8h*)&ov;
            } else {
                v4h ov;
                #pragma unroll
                for (int o = 0; o < OC; ++o) {
                    float v = acc[o];
                    if (RELU) v = fmaxf(v, 0.0f);
                    ov[o & 3] = (_Float16)v;
                }
                *(v4h*)&out[base] = *(v4h*)&ov;
            }
        } else {
            #pragma unroll
            for (int o = 0; o < OC; ++o) {
                float v = acc[o];
                if (RELU) v = fmaxf(v, 0.0f);
                out[((size_t)(b * OC + o) * H_ + h) * (size_t)WoutS + w] = (OutT)v;
            }
        }
    }
}

// ---------------------------------------------------------------------------
// Implicit-GEMM conv via v_wmma_f32_16x16x32_f16.
// Input is channel-interleaved [b][h][Win][IC]; K ordered kk = tap*IC + ch so
// every 8-half K-run is contiguous & aligned:
//   IC=8 (conv) : A fragment = 2x ds_load_b128
//   IC=4 (convT): A fragment = 4x ds_load_b64
// Window staged global->LDS with one contiguous run of async b128 copies
// (interior tiles); weights staged transposed [16][K] -> B = 2x ds_load_b128.
// Epilogue: one aligned global_store_b128 of 8 consecutive positions/lane.
// ---------------------------------------------------------------------------
template<int IC, int OC, int KW, bool TR, bool RELU>
__global__ __launch_bounds__(256)
void conv_wmma(const _Float16* __restrict__ in, const float* __restrict__ wg,
               const float* __restrict__ bias, _Float16* __restrict__ out,
               int Win, int Wout, int WoutS)
{
    constexpr int KD   = IC * KW;    // 512 or 256
    constexpr int WINC = 128 + KW;   // staged window positions (192)
    __shared__ __align__(16) _Float16 s_in[WINC * IC];
    __shared__ __align__(16) _Float16 s_Bt[16 * KD];

    const int tid = threadIdx.x;
    const int row = blockIdx.y;
    const int b = row / H_, h = row % H_;
    const int wb  = blockIdx.x * 128;
    const int wb0 = TR ? (wb - KW) : wb;
    const size_t rowIlv = ((size_t)b * H_ + h) * (size_t)Win * IC;

    // ---- stage weights transposed + zero-padded: s_Bt[n][kk], kk=tap*IC+ch
    for (int i = tid; i < 16 * KD; i += 256) {
        int n = i / KD, kk = i % KD;
        int ch = kk % IC, tap = kk / IC;
        float v = 0.0f;
        if (n < OC)
            v = TR ? wg[(ch * OC + n) * KW + tap] : wg[(n * IC + ch) * KW + tap];
        s_Bt[i] = (_Float16)v;
    }

    // ---- stage input window (one contiguous, 16B-aligned byte range) ----
    const bool interior = (wb0 >= 0) && (wb0 + WINC <= Win);
    if (interior) {
        constexpr int SEGS = WINC * IC / 8;  // 192 or 96 b128 copies
        if (tid < SEGS) {
            uint32_t lds = (uint32_t)(uintptr_t)&s_in[tid * 8];
            uint64_t ga  = (uint64_t)(uintptr_t)(in + rowIlv +
                               (size_t)wb0 * IC + (size_t)tid * 8);
            asm volatile("global_load_async_to_lds_b128 %0, %1, off"
                         :: "v"(lds), "v"(ga) : "memory");
        }
        asm volatile("s_wait_asynccnt 0x0" ::: "memory");
    } else {
        for (int i = tid; i < WINC * IC; i += 256) {
            int t = i / IC, ch = i % IC;
            int wsrc = wb0 + t;
            _Float16 v = (_Float16)0.0f;
            if (wsrc >= 0 && wsrc < Win)
                v = in[rowIlv + (size_t)wsrc * IC + ch];
            s_in[i] = v;
        }
    }
    __syncthreads();

    const int wave = tid >> 5, lane = tid & 31;
    const int l16 = lane & 15, hi = lane >> 4;
    const int koff = hi ? 8 : 0;
    const int mrow = wave * 16 + l16;
    const v8h* bpt = (const v8h*)&s_Bt[l16 * KD];

    v8f c = {0.f, 0.f, 0.f, 0.f, 0.f, 0.f, 0.f, 0.f};
    for (int kti = 0; kti < KD / 32; ++kti) {
        v16h a, bb;
        v8h blo = bpt[kti * 4 + hi];
        v8h bhi = bpt[kti * 4 + hi + 2];
        #pragma unroll
        for (int j = 0; j < 8; ++j) { bb[j] = blo[j]; bb[8 + j] = bhi[j]; }
        if (IC == 8) {           // conv: run = 8 channels at one tap (b128)
            const v8h* sp = (const v8h*)s_in;
            int tap = (kti * 32 + koff) >> 3;
            v8h lo = sp[mrow + tap];
            v8h hh = sp[mrow + tap + 2];
            #pragma unroll
            for (int j = 0; j < 8; ++j) { a[j] = lo[j]; a[8 + j] = hh[j]; }
        } else {                 // IC==4, convT: run = 2 taps x 4 channels (b64)
            const v4h* sp4 = (const v4h*)s_in;
            int iA = mrow + KW - ((kti * 32 + koff) >> 2);
            v4h g0 = sp4[iA],     g1 = sp4[iA - 1];
            v4h g2 = sp4[iA - 4], g3 = sp4[iA - 5];
            #pragma unroll
            for (int j = 0; j < 4; ++j) {
                a[j] = g0[j]; a[4 + j] = g1[j]; a[8 + j] = g2[j]; a[12 + j] = g3[j];
            }
        }
        c = __builtin_amdgcn_wmma_f32_16x16x32_f16(false, a, false, bb,
                                                   (short)0, c, false, false);
    }

    if (l16 < OC) {
        float bv = bias[l16];
        v8h ov;
        #pragma unroll
        for (int r = 0; r < 8; ++r) {
            float v = c[r] + bv;
            if (RELU) v = fmaxf(v, 0.0f);
            ov[r] = (_Float16)v;
        }
        const int wstart = wb + wave * 16 + hi * 8;   // 8 consecutive positions
        size_t base = ((size_t)(b * OC + l16) * H_ + h) * (size_t)WoutS;
        if (wstart + 8 <= WoutS) {
            *(v8h*)&out[base + wstart] = ov;          // aligned b128 store
        } else {
            #pragma unroll
            for (int r = 0; r < 8; ++r)
                if (wstart + r < Wout) out[base + wstart + r] = ov[r];
        }
    }
}

// ---------------------------------------------------------------------------
// conv3 (4->1, kw=8) writing the latent TRANSPOSED: z3t[b][w][h] (f32 + f16),
// making the VQ GEMM A-operand contiguous (16B-aligned groups) along K=h.
// ---------------------------------------------------------------------------
__global__ __launch_bounds__(256)
void conv3_kernel(const _Float16* __restrict__ z2, const float* __restrict__ w3,
                  const float* __restrict__ b3, float* __restrict__ z3t,
                  _Float16* __restrict__ z3t_h)
{
    constexpr int WIN = 256 + 7;
    __shared__ float s_in[4 * WIN];
    __shared__ float s_w[32];

    const int tid = threadIdx.x;
    const int row = blockIdx.y;
    const int b = row / H_, h = row % H_;
    const int wb = blockIdx.x * 256;

    for (int i = tid; i < 4 * WIN; i += 256) {
        int ch = i / WIN, t = i % WIN;
        int wsrc = wb + t;
        float v = 0.0f;
        if (wsrc < W2_)
            v = (float)z2[((size_t)(b * 4 + ch) * H_ + h) * (size_t)W2S_ + wsrc];
        s_in[i] = v;
    }
    if (tid < 32) s_w[tid] = w3[tid];
    __syncthreads();

    const int w = wb + tid;
    if (w < W3_) {
        float acc = b3[0];
        #pragma unroll
        for (int ch = 0; ch < 4; ++ch)
            #pragma unroll
            for (int tap = 0; tap < 8; ++tap)
                acc = fmaf(s_in[ch * WIN + tid + tap], s_w[ch * 8 + tap], acc);
        size_t o = ((size_t)b * W3_ + w) * DEMB + h;
        z3t[o]   = acc;
        z3t_h[o] = (_Float16)acc;
    }
}

// ---------------------------------------------------------------------------
// Codebook prep: f16 copy + ||e||^2 per code.
// ---------------------------------------------------------------------------
__global__ __launch_bounds__(128)
void emb_prep(const float* __restrict__ emb, _Float16* __restrict__ emb_h,
              float* __restrict__ enorm)
{
    __shared__ float red[128];
    int e = blockIdx.x, h = threadIdx.x;
    float v = emb[e * DEMB + h];
    emb_h[e * DEMB + h] = (_Float16)v;
    red[h] = v * v;
    __syncthreads();
    for (int s = 64; s > 0; s >>= 1) { if (h < s) red[h] += red[h + s]; __syncthreads(); }
    if (h == 0) enorm[e] = red[0];
}

// ---------------------------------------------------------------------------
// VQ nearest-code search: WMMA GEMM (M=rows, N=512 codes, K=128).
// Two M-tiles per wave: each B fragment feeds two independent WMMA chains.
// argmin of (||e||^2 - 2 x.e) with per-lane running min + shfl reduction.
// ---------------------------------------------------------------------------
__global__ __launch_bounds__(256)
void vq_argmin(const _Float16* __restrict__ xh, const _Float16* __restrict__ eh,
               const float* __restrict__ enorm, int* __restrict__ idx)
{
    const int tid = threadIdx.x, wave = tid >> 5, lane = tid & 31;
    const int l16 = lane & 15, hi = lane >> 4;
    const int nBase = blockIdx.x * 256 + wave * 32;

    v16h a[2][4];
    #pragma unroll
    for (int mt = 0; mt < 2; ++mt) {
        int rowA = nBase + mt * 16 + l16;
        if (rowA >= NROWS) rowA = NROWS - 1;
        const v8h* ap = (const v8h*)(xh + (size_t)rowA * DEMB);
        #pragma unroll
        for (int kt = 0; kt < 4; ++kt) {
            v8h lo = ap[kt * 4 + hi];
            v8h hh = ap[kt * 4 + hi + 2];
            #pragma unroll
            for (int j = 0; j < 8; ++j) { a[mt][kt][j] = lo[j]; a[mt][kt][8 + j] = hh[j]; }
        }
    }

    float best[2][8]; int bidx[2][8];
    #pragma unroll
    for (int mt = 0; mt < 2; ++mt)
        #pragma unroll
        for (int r = 0; r < 8; ++r) { best[mt][r] = 3.4e38f; bidx[mt][r] = 0; }

    for (int et = 0; et < NEMB / 16; ++et) {
        int e = et * 16 + l16;
        const v8h* bp = (const v8h*)(eh + (size_t)e * DEMB);
        v8f c0 = {0.f, 0.f, 0.f, 0.f, 0.f, 0.f, 0.f, 0.f};
        v8f c1 = {0.f, 0.f, 0.f, 0.f, 0.f, 0.f, 0.f, 0.f};
        #pragma unroll
        for (int kt = 0; kt < 4; ++kt) {
            v16h bb;
            v8h lo = bp[kt * 4 + hi];
            v8h hh = bp[kt * 4 + hi + 2];
            #pragma unroll
            for (int j = 0; j < 8; ++j) { bb[j] = lo[j]; bb[8 + j] = hh[j]; }
            c0 = __builtin_amdgcn_wmma_f32_16x16x32_f16(false, a[0][kt], false, bb,
                                                        (short)0, c0, false, false);
            c1 = __builtin_amdgcn_wmma_f32_16x16x32_f16(false, a[1][kt], false, bb,
                                                        (short)0, c1, false, false);
        }
        float en = enorm[e];
        #pragma unroll
        for (int r = 0; r < 8; ++r) {
            float d0 = en - 2.0f * c0[r];
            if (d0 < best[0][r]) { best[0][r] = d0; bidx[0][r] = e; }
            float d1 = en - 2.0f * c1[r];
            if (d1 < best[1][r]) { best[1][r] = d1; bidx[1][r] = e; }
        }
    }

    #pragma unroll
    for (int mt = 0; mt < 2; ++mt) {
        #pragma unroll
        for (int r = 0; r < 8; ++r) {
            float v = best[mt][r]; int ix = bidx[mt][r];
            for (int off = 8; off > 0; off >>= 1) {
                float ov = __shfl_xor(v, off, 16);
                int   oi = __shfl_xor(ix, off, 16);
                if (ov < v || (ov == v && oi < ix)) { v = ov; ix = oi; }
            }
            if (l16 == 0) {
                int rowD = nBase + mt * 16 + r + hi * 8;
                if (rowD < NROWS) idx[rowD] = ix;
            }
        }
    }
}

// ---------------------------------------------------------------------------
// Gather codes -> qz (f16, [b][h][w] layout for decoder), accumulate SSE and
// the code-usage histogram.
// ---------------------------------------------------------------------------
__global__ __launch_bounds__(128)
void vq_apply(const int* __restrict__ idx, const float* __restrict__ emb,
              const float* __restrict__ z3t, _Float16* __restrict__ qz,
              float* __restrict__ sse, int* __restrict__ hist)
{
    __shared__ float red[128];
    const int n = blockIdx.x, h = threadIdx.x;
    const int e = idx[n];
    const int b = n / W3_, w = n % W3_;
    float q = emb[e * DEMB + h];
    float x = z3t[(size_t)n * DEMB + h];
    float d = (q - x) * (q - x);
    qz[((size_t)b * H_ + h) * (size_t)W3_ + w] = (_Float16)q;
    red[h] = d;
    __syncthreads();
    for (int s = 64; s > 0; s >>= 1) { if (h < s) red[h] += red[h + s]; __syncthreads(); }
    if (h == 0) { atomicAdd(sse, red[0]); atomicAdd(hist + e, 1); }
}

// ---------------------------------------------------------------------------
// Final scalars: loss = 1.5 * MSE ; perplexity = exp(-sum p log(p+1e-10)).
// ---------------------------------------------------------------------------
__global__ __launch_bounds__(512)
void finalize(const float* __restrict__ sse, const int* __restrict__ hist,
              float* __restrict__ d_out)
{
    __shared__ float red[512];
    int t = threadIdx.x;
    float p = (float)hist[t] / (float)NROWS;
    red[t] = p * logf(p + 1e-10f);
    __syncthreads();
    for (int s = 256; s > 0; s >>= 1) { if (t < s) red[t] += red[t + s]; __syncthreads(); }
    if (t == 0) {
        d_out[0] = (*sse) / (float)((size_t)NROWS * DEMB) * 1.5f;
        d_out[1 + YSIZE] = expf(-red[0]);
    }
}

// ---------------------------------------------------------------------------
extern "C" void kernel_launch(void* const* d_in, const int* in_sizes, int n_in,
                              void* d_out, int out_size, void* d_ws, size_t ws_size,
                              hipStream_t stream)
{
    const float* x   = (const float*)d_in[0];
    const float* w1  = (const float*)d_in[1];
    const float* b1  = (const float*)d_in[2];
    const float* w2  = (const float*)d_in[3];
    const float* b2  = (const float*)d_in[4];
    const float* w3  = (const float*)d_in[5];
    const float* b3  = (const float*)d_in[6];
    const float* emb = (const float*)d_in[7];
    const float* wt1 = (const float*)d_in[8];
    const float* bt1 = (const float*)d_in[9];
    const float* wt2 = (const float*)d_in[10];
    const float* bt2 = (const float*)d_in[11];
    const float* wt3 = (const float*)d_in[12];
    const float* bt3 = (const float*)d_in[13];
    float* out = (float*)d_out;

    char* p = (char*)d_ws;
    auto alloc = [&](size_t bytes) -> void* {
        void* r = (void*)p;
        p += (bytes + 255) & ~(size_t)255;
        return r;
    };
    _Float16* z1i   = (_Float16*)alloc(sizeof(_Float16) * (size_t)B_ * H_ * W1_ * 8);
    _Float16* z2    = (_Float16*)alloc(sizeof(_Float16) * (size_t)B_ * 4 * H_ * W2S_);
    float*    z3t   = (float*)   alloc(sizeof(float)    * (size_t)NROWS * DEMB);
    _Float16* z3t_h = (_Float16*)alloc(sizeof(_Float16) * (size_t)NROWS * DEMB);
    _Float16* emb_h = (_Float16*)alloc(sizeof(_Float16) * NEMB * DEMB);
    float*    enorm = (float*)   alloc(sizeof(float) * NEMB);
    int*      idxb  = (int*)     alloc(sizeof(int) * NROWS);
    _Float16* qz    = (_Float16*)alloc(sizeof(_Float16) * (size_t)NROWS * DEMB);
    _Float16* y1i   = (_Float16*)alloc(sizeof(_Float16) * (size_t)B_ * H_ * W2_ * 4);
    _Float16* y2    = (_Float16*)alloc(sizeof(_Float16) * (size_t)B_ * 8 * H_ * W1S_);
    float*    sse   = (float*)   alloc(sizeof(float));
    int*      hist  = (int*)     alloc(sizeof(int) * NEMB);

    const int ROWS = B_ * H_;  // 4096 (b,h) rows

    init_accum<<<1, 512, 0, stream>>>(sse, hist);

    // Encoder
    conv_direct<float, _Float16, 1, 8, 32, false, true, true>
        <<<dim3((W1_ + 255) / 256, ROWS), 256, 0, stream>>>(x, w1, b1, z1i,
                                                            W0_, W0_, W1_, W1_);
    conv_wmma<8, 4, 64, false, true>
        <<<dim3((W2_ + 127) / 128, ROWS), 256, 0, stream>>>(z1i, w2, b2, z2,
                                                            W1_, W2_, W2S_);
    conv3_kernel<<<dim3((W3_ + 255) / 256, ROWS), 256, 0, stream>>>(z2, w3, b3, z3t, z3t_h);

    // Vector quantizer
    emb_prep<<<NEMB, 128, 0, stream>>>(emb, emb_h, enorm);
    vq_argmin<<<(NROWS + 255) / 256, 256, 0, stream>>>(z3t_h, emb_h, enorm, idxb);
    vq_apply<<<NROWS, 128, 0, stream>>>(idxb, emb, z3t, qz, sse, hist);
    finalize<<<1, 512, 0, stream>>>(sse, hist, out);

    // Decoder (transposed convs = full convs)
    conv_direct<_Float16, _Float16, 1, 4, 8, true, true, true>
        <<<dim3((W2_ + 255) / 256, ROWS), 256, 0, stream>>>(qz, wt1, bt1, y1i,
                                                            W3_, W3_, W2_, W2_);
    conv_wmma<4, 8, 64, true, true>
        <<<dim3((W1_ + 127) / 128, ROWS), 256, 0, stream>>>(y1i, wt2, bt2, y2,
                                                            W2_, W1_, W1S_);
    conv_direct<_Float16, float, 8, 1, 32, true, false, false>
        <<<dim3((W0_ + 255) / 256, ROWS), 256, 0, stream>>>(y2, wt3, bt3, out + 1,
                                                            W1_, W1S_, W0_, W0_);
}
